// GradEnergyMessagePassing_46196668236124
// MI455X (gfx1250) — matrix operand order repaired and checked by reading, not compiled, not run on hardware
//
#include <hip/hip_runtime.h>
#include <hip/hip_bf16.h>

typedef _Float16 v8h  __attribute__((ext_vector_type(8)));
typedef _Float16 v16h __attribute__((ext_vector_type(16)));
typedef float    v8f  __attribute__((ext_vector_type(8)));

#define D_FEAT  64
#define HIDDEN  128
#define TILE_E  32

// silu via hardware v_rcp_f32 (avoids precise-division expansion)
__device__ __forceinline__ float siluf(float v) {
    return v * __builtin_amdgcn_rcpf(1.0f + __expf(-v));
}

__device__ __forceinline__ v16h load_afrag(const _Float16* p) {
    v8h lo = *(const v8h*)p;
    v8h hi = *(const v8h*)(p + 16);
    return __builtin_shufflevector(lo, hi, 0, 1, 2, 3, 4, 5, 6, 7,
                                   8, 9, 10, 11, 12, 13, 14, 15);
}

// ---------------- zero the output buffer (harness poisons it) ----------------
__global__ void zero_out_kernel(float* __restrict__ out, int n) {
    for (int i = blockIdx.x * blockDim.x + threadIdx.x; i < n;
         i += gridDim.x * blockDim.x)
        out[i] = 0.0f;
}

// ---- convert + transpose weights to f16, wh[n*128+k] = (f16)W[k*128+n] -----
__global__ void prep_weights_kernel(const float* __restrict__ W1,
                                    const float* __restrict__ W2,
                                    _Float16* __restrict__ w1h,
                                    _Float16* __restrict__ w2h) {
    for (int i = blockIdx.x * blockDim.x + threadIdx.x; i < HIDDEN * HIDDEN;
         i += gridDim.x * blockDim.x) {
        int k = i >> 7;
        int n = i & 127;
        w1h[n * HIDDEN + k] = (_Float16)W1[i];
        w2h[n * HIDDEN + k] = (_Float16)W2[i];
    }
}

// ---- f16 mirror of node features (halves gather traffic, kills converts) ---
__global__ void prep_x_kernel(const float* __restrict__ x,
                              _Float16* __restrict__ xh, int n) {
    for (int i = blockIdx.x * blockDim.x + threadIdx.x; i < n;
         i += gridDim.x * blockDim.x)
        xh[i] = (_Float16)x[i];
}

// ---------------------------- main fused kernel ------------------------------
__global__ void __launch_bounds__(256)
ebm_mp_kernel(const float* __restrict__ x,
              const _Float16* __restrict__ xh,    // f16 mirror (may be unused)
              const long long* __restrict__ ei,   // [2][nEdges] int64
              const float* __restrict__ t,
              const float* __restrict__ b1,
              const float* __restrict__ Wt,
              const float* __restrict__ bt,
              const float* __restrict__ b2,
              const _Float16* __restrict__ w1h,   // [n][k] f16
              const _Float16* __restrict__ w2h,   // [n][k] f16
              float* __restrict__ out,            // [N_NODES][64]
              int nEdges, int fullTiles, int useF16X)
{
    __shared__ alignas(16) _Float16 sA[TILE_E][2 * D_FEAT]; // edge_input f16
    __shared__ alignas(16) float    sT[TILE_E][HIDDEN];     // silu(t*Wt+bt)+b1
    __shared__ alignas(16) _Float16 sH[TILE_E][HIDDEN];     // hidden f16
    __shared__ int sIdx[2][TILE_E];                         // row / col ids

    const int tid   = threadIdx.x;
    const int lane  = tid & 31;
    const int wv    = tid >> 5;           // 0..7 waves
    const int nIdx  = wv * 16 + (lane & 15);
    const int khalf = (lane >> 4) * 8;    // B-layout: lanes 16-31 hold K+8..15, K+24..31

    // ---- register-resident weight fragments (B matrices), loaded once ----
    v16h B1[4], B2[4];
#pragma unroll
    for (int ks = 0; ks < 4; ++ks) {
        B1[ks] = load_afrag(w1h + nIdx * HIDDEN + ks * 32 + khalf);
        B2[ks] = load_afrag(w2h + nIdx * HIDDEN + ks * 32 + khalf);
    }

    const int eLoc  = tid >> 4;           // 0..15 (edge pair base)
    const int part  = tid & 15;           // 16 threads per edge, 8 cols each
    const int cbase = part * 8;           // 0..120

    // ---- loop-invariant small vectors for the time-embedding epilogue ----
    float wtv[8], btb[8], b1v[8];
#pragma unroll
    for (int j = 0; j < 8; ++j) {
        int c = cbase + j;
        wtv[j] = Wt[c];
        btb[j] = bt[c];
        b1v[j] = b1[c];
    }
    const float bias2 = b2[nIdx];         // loop-invariant scatter bias
    const int   side  = (nIdx >= D_FEAT) ? 1 : 0;   // wave-uniform
    const int   oc    = nIdx - side * D_FEAT;

    const int eiSideOff = (part >= 8) ? nEdges : 0;
    const int xcol = cbase & 63;

    const int aM   = lane & 15;           // A-layout row
    const int aOff = (lane >> 4) * 8;     // A-layout: lanes 16-31 hold K+8..

    for (int tb = blockIdx.x; tb < fullTiles; tb += gridDim.x) {
        // ---------------- gather stage (all 256 threads, guard-free) --------
#pragma unroll
        for (int half = 0; half < 2; ++half) {
            const int el = eLoc + half * 16;
            const int e  = tb * TILE_E + el;
            // node ids < 2^31: keep all derived offsets in 32-bit for GVS
            const int node = (int)ei[eiSideOff + e];
            v8h h8;
            if (useF16X) {
                h8 = *(const v8h*)(xh + node * D_FEAT + xcol);
            } else {
                const float4* px = (const float4*)(x + node * D_FEAT + xcol);
                float4 f0 = px[0];
                float4 f1 = px[1];
                h8 = (v8h){(_Float16)f0.x, (_Float16)f0.y,
                           (_Float16)f0.z, (_Float16)f0.w,
                           (_Float16)f1.x, (_Float16)f1.y,
                           (_Float16)f1.z, (_Float16)f1.w};
            }
            *(v8h*)&sA[el][cbase] = h8;

            const float tv = t[e];
            float4 t0, t1;
            t0.x = siluf(fmaf(tv, wtv[0], btb[0])) + b1v[0];
            t0.y = siluf(fmaf(tv, wtv[1], btb[1])) + b1v[1];
            t0.z = siluf(fmaf(tv, wtv[2], btb[2])) + b1v[2];
            t0.w = siluf(fmaf(tv, wtv[3], btb[3])) + b1v[3];
            t1.x = siluf(fmaf(tv, wtv[4], btb[4])) + b1v[4];
            t1.y = siluf(fmaf(tv, wtv[5], btb[5])) + b1v[5];
            t1.z = siluf(fmaf(tv, wtv[6], btb[6])) + b1v[6];
            t1.w = siluf(fmaf(tv, wtv[7], btb[7])) + b1v[7];
            *(float4*)&sT[el][cbase]     = t0;
            *(float4*)&sT[el][cbase + 4] = t1;
        }
        if (tid < 64) {
            int s = tid >> 5, m = tid & 31;
            sIdx[s][m] = (int)ei[(long long)s * nEdges + tb * TILE_E + m];
        }
        __syncthreads();

        // ---------------- GEMM1: h_pre = edge_input @ W1 ----------------
        v8f acc0 = {}, acc1 = {};
#pragma unroll
        for (int ks = 0; ks < 4; ++ks) {
            v16h a0 = load_afrag(&sA[aM][ks * 32 + aOff]);
            v16h a1 = load_afrag(&sA[aM + 16][ks * 32 + aOff]);
            acc0 = __builtin_amdgcn_wmma_f32_16x16x32_f16(
                false, a0, false, B1[ks], (short)0, acc0, false, false);
            acc1 = __builtin_amdgcn_wmma_f32_16x16x32_f16(
                false, a1, false, B1[ks], (short)0, acc1, false, false);
        }
        // epilogue: + (t_emb + b1), silu, store hidden as f16
#pragma unroll
        for (int r = 0; r < 8; ++r) {
            int M0 = r + 8 * (lane >> 4);
            float p0 = acc0[r] + sT[M0][nIdx];
            float p1 = acc1[r] + sT[M0 + 16][nIdx];
            sH[M0][nIdx]      = (_Float16)siluf(p0);
            sH[M0 + 16][nIdx] = (_Float16)siluf(p1);
        }
        __syncthreads();

        // ---------------- GEMM2: edge_out = h @ W2 ----------------
        v8f d0 = {}, d1 = {};
#pragma unroll
        for (int ks = 0; ks < 4; ++ks) {
            v16h a0 = load_afrag(&sH[aM][ks * 32 + aOff]);
            v16h a1 = load_afrag(&sH[aM + 16][ks * 32 + aOff]);
            d0 = __builtin_amdgcn_wmma_f32_16x16x32_f16(
                false, a0, false, B2[ks], (short)0, d0, false, false);
            d1 = __builtin_amdgcn_wmma_f32_16x16x32_f16(
                false, a1, false, B2[ks], (short)0, d1, false, false);
        }

        // ---------------- scatter (segment_sum via atomics) ----------------
        // side/oc are wave-uniform: waves 0-3 -> row side, 4-7 -> col side.
        // 32-bit element offsets (max 3.2M) -> GVS addressing for atomics.
#pragma unroll
        for (int r = 0; r < 8; ++r) {
            int M0 = r + 8 * (lane >> 4);
            int off0 = sIdx[side][M0]      * D_FEAT + oc;
            int off1 = sIdx[side][M0 + 16] * D_FEAT + oc;
            atomicAdd(out + off0, d0[r] + bias2);
            atomicAdd(out + off1, d1[r] + bias2);
        }
        __syncthreads();  // protect sA/sT/sIdx before next tile's gather
    }
}

// ---- scalar tail for nEdges % TILE_E (never used when divisible) -----------
__global__ void edge_tail_kernel(const float* __restrict__ x,
                                 const long long* __restrict__ ei,
                                 const float* __restrict__ t,
                                 const float* __restrict__ W1,
                                 const float* __restrict__ b1,
                                 const float* __restrict__ Wt,
                                 const float* __restrict__ bt,
                                 const float* __restrict__ W2,
                                 const float* __restrict__ b2,
                                 float* __restrict__ out,
                                 int nEdges, int startEdge)
{
    const int e = startEdge + blockIdx.x;
    if (e >= nEdges) return;
    __shared__ float sIn[2 * D_FEAT];
    __shared__ float sHid[HIDDEN];
    const int j = threadIdx.x;  // 0..127
    const int r = (int)ei[e];
    const int c = (int)ei[(long long)nEdges + e];
    sIn[j] = (j < D_FEAT) ? x[r * D_FEAT + j] : x[c * D_FEAT + (j - D_FEAT)];
    __syncthreads();
    const float tv = t[e];
    float acc = siluf(fmaf(tv, Wt[j], bt[j])) + b1[j];
    for (int k = 0; k < 2 * D_FEAT; ++k)
        acc = fmaf(sIn[k], W1[k * HIDDEN + j], acc);
    sHid[j] = siluf(acc);
    __syncthreads();
    float o = b2[j];
    for (int k = 0; k < HIDDEN; ++k)
        o = fmaf(sHid[k], W2[k * 2 * D_FEAT + j], o);
    const int node = (j < D_FEAT) ? r : c;
    atomicAdd(out + node * D_FEAT + (j & (D_FEAT - 1)), o);
}

// ------------------------------- launcher ------------------------------------
extern "C" void kernel_launch(void* const* d_in, const int* in_sizes, int n_in,
                              void* d_out, int out_size, void* d_ws,
                              size_t ws_size, hipStream_t stream) {
    const float*     x  = (const float*)d_in[0];
    const long long* ei = (const long long*)d_in[1];
    const float*     t  = (const float*)d_in[2];
    const float*     W1 = (const float*)d_in[3];
    const float*     b1 = (const float*)d_in[4];
    const float*     Wt = (const float*)d_in[5];
    const float*     bt = (const float*)d_in[6];
    const float*     W2 = (const float*)d_in[7];
    const float*     b2 = (const float*)d_in[8];
    float* out = (float*)d_out;

    const int nEdges = in_sizes[2];        // t has one entry per edge
    const int nXElem = in_sizes[0];        // nodes * 64

    _Float16* w1h = (_Float16*)d_ws;
    _Float16* w2h = w1h + HIDDEN * HIDDEN;
    _Float16* xh  = w2h + HIDDEN * HIDDEN;

    const size_t needBytes =
        (size_t)(2 * HIDDEN * HIDDEN + nXElem) * sizeof(_Float16);
    const int useF16X = (ws_size >= needBytes) ? 1 : 0;

    zero_out_kernel<<<4096, 256, 0, stream>>>(out, out_size);
    prep_weights_kernel<<<64, 256, 0, stream>>>(W1, W2, w1h, w2h);
    if (useF16X)
        prep_x_kernel<<<2048, 256, 0, stream>>>(x, xh, nXElem);

    const int fullTiles = nEdges / TILE_E;
    if (fullTiles > 0) {
        int grid = fullTiles < 8192 ? fullTiles : 8192;
        ebm_mp_kernel<<<grid, 256, 0, stream>>>(x, xh, ei, t, b1, Wt, bt, b2,
                                                w1h, w2h, out, nEdges,
                                                fullTiles, useF16X);
    }
    const int rem = nEdges - fullTiles * TILE_E;
    if (rem > 0)
        edge_tail_kernel<<<rem, HIDDEN, 0, stream>>>(
            x, ei, t, W1, b1, Wt, bt, W2, b2, out, nEdges,
            fullTiles * TILE_E);
}
